// ConstantTimeStrideAttentionFast_70635032150395
// MI455X (gfx1250) — compile-verified, matrix operand order checked by reading, not run on hardware
//
#include <hip/hip_runtime.h>
#include <hip/hip_bf16.h>
#include <math.h>

typedef __bf16 bf16;
typedef __attribute__((ext_vector_type(16))) __bf16 v16bf;
typedef __attribute__((ext_vector_type(8)))  __bf16 v8bf;
typedef __attribute__((ext_vector_type(4)))  __bf16 v4bf;
typedef __attribute__((ext_vector_type(8)))  float  v8f;

// 16-byte payload type the async-LDS builtins expect (int __vector(4))
typedef int b128_t __attribute__((vector_size(16)));
typedef __attribute__((address_space(1))) b128_t glob_b128;
typedef __attribute__((address_space(3))) b128_t lds_b128;

// Problem constants (from reference setup_inputs)
#define BATCH 2
#define SEQ   4096
#define DMODEL 1536
#define NHEAD 12
#define DHEAD 128
#define QKVN  4608           // 3*DMODEL
#define MROWS 8192           // BATCH*SEQ

// LDS tile geometry: 128 rows x 32 bf16 (64B), padded to 80B row stride
#define TILE_ROWS   128
#define TILE_PITCH  80        // 16B-aligned, avoids 4-way ds bank conflicts
#define TILE_BYTES  (TILE_ROWS * TILE_PITCH)   // 10240
#define BUF_STRIDE  (2 * TILE_BYTES)           // A+B panel per phase

// ---------------------------------------------------------------------------
// Async global->LDS copy (CDNA5 ASYNCcnt path), guarded with fallback
// ---------------------------------------------------------------------------
#if __has_builtin(__builtin_amdgcn_global_load_async_to_lds_b128) && \
    __has_builtin(__builtin_amdgcn_s_wait_asynccnt)
#define USE_ASYNC_LDS 1
#else
#define USE_ASYNC_LDS 0
#endif

__device__ __forceinline__ void copy16(const bf16* g, char* l) {
#if USE_ASYNC_LDS
    __builtin_amdgcn_global_load_async_to_lds_b128(
        (glob_b128*)(g), (lds_b128*)(l), 0, 0);
#else
    *reinterpret_cast<uint4*>(l) = *reinterpret_cast<const uint4*>(g);
#endif
}

__device__ __forceinline__ void copy_join() {
#if USE_ASYNC_LDS
    __builtin_amdgcn_s_wait_asynccnt(0);
#endif
}

// ---------------------------------------------------------------------------
// fp32 -> bf16 conversion (vectorized x4)
// ---------------------------------------------------------------------------
__global__ void f32_to_bf16_x4(const float* __restrict__ in,
                               bf16* __restrict__ out, int n4) {
    int i = blockIdx.x * blockDim.x + threadIdx.x;
    int stride = gridDim.x * blockDim.x;
    for (; i < n4; i += stride) {
        float4 f = reinterpret_cast<const float4*>(in)[i];
        v4bf o;
        o[0] = (bf16)f.x; o[1] = (bf16)f.y; o[2] = (bf16)f.z; o[3] = (bf16)f.w;
        reinterpret_cast<v4bf*>(out)[i] = o;
    }
}

// ---------------------------------------------------------------------------
// WMMA bf16 GEMM:  C[M,N] = A[M,K] * W[N,K]^T + bias[N]
//   block: 256 threads = 8 waves, block tile 128(M) x 128(N)
//   wave tile: 64(M) x 32(N) = 4x2 v_wmma_f32_16x16x32_bf16 accumulators
//   A/B K-panels (128x32 bf16 each) double-buffered in LDS via async copies.
//   M,N,K all multiples of the tiles for our shapes (no bounds checks).
// ---------------------------------------------------------------------------
__device__ __forceinline__ v8f wmma_bf16(v16bf a, v16bf b, v8f c) {
    return __builtin_amdgcn_wmma_f32_16x16x32_bf16(
        /*neg_a=*/false, a, /*neg_b=*/false, b,
        /*c_mod=*/(short)0, c, /*reuse_a=*/false, /*reuse_b=*/false);
}

// Fragment from LDS tile: lane's 16 bf16 = K {h*8..h*8+7, 16+h*8..16+h*8+7}
__device__ __forceinline__ v16bf lds_frag(const char* base, int row, int half) {
    union { v16bf v; v8bf h[2]; } u;
    const char* p = base + row * TILE_PITCH + half * 16;
    u.h[0] = *reinterpret_cast<const v8bf*>(p);
    u.h[1] = *reinterpret_cast<const v8bf*>(p + 32);
    return u.v;
}

// Stage one 128x32 K-panel pair (A rows + W rows) into LDS. 1024 16B
// segments, 4 per thread. Row r occupies bytes [r*80, r*80+64).
__device__ __forceinline__ void issue_tile_copy(
    const bf16* __restrict__ Ablk, const bf16* __restrict__ Wblk,
    int K, int k0, char* ldsA, char* ldsB, int t) {
#pragma unroll
    for (int i = 0; i < 2; ++i) {
        const int s = t + i * 256;           // 0..511
        const int row = s >> 2, seg = s & 3; // 4 x 16B per 64B row
        copy16(Ablk + (size_t)row * K + k0 + seg * 8,
               ldsA + row * TILE_PITCH + seg * 16);
    }
#pragma unroll
    for (int i = 0; i < 2; ++i) {
        const int s = t + i * 256;
        const int row = s >> 2, seg = s & 3;
        copy16(Wblk + (size_t)row * K + k0 + seg * 8,
               ldsB + row * TILE_PITCH + seg * 16);
    }
}

template <typename OutT>
__global__ __launch_bounds__(256) void gemm_bf16_wmma(
    const bf16* __restrict__ A,     // [M,K] row-major
    const bf16* __restrict__ W,     // [N,K] row-major (acts as B^T)
    const float* __restrict__ bias, // [N]
    OutT* __restrict__ C,           // [M,N] row-major
    int M, int N, int K) {
    __shared__ char smem[2 * BUF_STRIDE];   // double-buffered A+B = 40KB

    const int tid  = threadIdx.x;
    const int lane = tid & 31;
    const int wave = tid >> 5;
    const int wm   = wave >> 2;          // 0..1  (M direction)
    const int wn   = wave & 3;           // 0..3  (N direction)
    const int r    = lane & 15;
    const int half = lane >> 4;
    const int row0 = blockIdx.y * 128 + wm * 64;
    const int col0 = blockIdx.x * 128 + wn * 32;

    const bf16* Ablk = A + (size_t)(blockIdx.y * 128) * K;
    const bf16* Wblk = W + (size_t)(blockIdx.x * 128) * K;

    v8f zero = {};
    v8f acc[4][2];
#pragma unroll
    for (int i = 0; i < 4; ++i)
#pragma unroll
        for (int j = 0; j < 2; ++j) acc[i][j] = zero;

    const int nk = K / 32;
    // prologue: stage tile 0 into phase-0 buffers
    issue_tile_copy(Ablk, Wblk, K, 0, smem, smem + TILE_BYTES, tid);

    for (int kt = 0; kt < nk; ++kt) {
        copy_join();            // s_wait_asynccnt 0
        __syncthreads();        // all waves' copies of this phase visible
        if (kt + 1 < nk) {
            char* nb = smem + ((kt + 1) & 1) * BUF_STRIDE;
            issue_tile_copy(Ablk, Wblk, K, (kt + 1) * 32,
                            nb, nb + TILE_BYTES, tid);
        }

        const char* lA = smem + (kt & 1) * BUF_STRIDE;
        const char* lB = lA + TILE_BYTES;
        v16bf af[4], bfr[2];
#pragma unroll
        for (int i = 0; i < 4; ++i)
            af[i] = lds_frag(lA, wm * 64 + i * 16 + r, half);
#pragma unroll
        for (int j = 0; j < 2; ++j)
            bfr[j] = lds_frag(lB, wn * 32 + j * 16 + r, half);
#pragma unroll
        for (int i = 0; i < 4; ++i)
#pragma unroll
            for (int j = 0; j < 2; ++j)
                acc[i][j] = wmma_bf16(af[i], bfr[j], acc[i][j]);
        // no trailing barrier: next iteration's wait+barrier precedes the
        // next copy that could overwrite this buffer
    }

    // C/D layout: VGPR e -> row (subtile + half*8 + e), col = col0 + (lane&15)
#pragma unroll
    for (int i = 0; i < 4; ++i) {
#pragma unroll
        for (int j = 0; j < 2; ++j) {
            const int rowB = row0 + i * 16 + half * 8;
            const int colB = col0 + j * 16 + r;
            const float bv = bias[colB];
            OutT* p = C + (size_t)rowB * N + colB;
#pragma unroll
            for (int e = 0; e < 8; ++e)
                p[(size_t)e * N] = (OutT)(acc[i][j][e] + bv);
        }
    }
}

// ---------------------------------------------------------------------------
// Anchor attention: one wave per (b, h, s). 12 anchors, d=128 (4 per lane).
// qkv layout: row = b*SEQ+s (stride QKVN), q at col 0, k at 1536, v at 3072,
// head h occupies cols h*128..h*128+127 within each section.
// ---------------------------------------------------------------------------
__global__ __launch_bounds__(256) void anchor_attn(
    const bf16* __restrict__ qkv, const float* __restrict__ gs,
    bf16* __restrict__ ctx) {
    const int lane = threadIdx.x & 31;
    const int wid  = blockIdx.x * 8 + (threadIdx.x >> 5);  // 0..B*H*S-1
    const int s    = wid & (SEQ - 1);
    const int t    = wid >> 12;           // SEQ = 2^12
    const int h    = t % NHEAD;
    const int b    = t / NHEAD;

    // group softmax -> per-group log-weights
    const float g0 = gs[0], g1 = gs[1], g2 = gs[2];
    const float mg = fmaxf(g0, fmaxf(g1, g2));
    const float lse = __logf(__expf(g0 - mg) + __expf(g1 - mg) + __expf(g2 - mg));
    const float lw0 = g0 - mg - lse, lw1 = g1 - mg - lse, lw2 = g2 - mg - lse;

    // anchors
    int anc[12];
    const int offs[10] = {-3, -2, -1, 1, 2, 3, -10, -5, 5, 10};
#pragma unroll
    for (int a = 0; a < 10; ++a) {
        int v = s + offs[a];
        anc[a] = v < 0 ? 0 : (v > SEQ - 1 ? SEQ - 1 : v);
    }
    anc[10] = 0;
    anc[11] = SEQ - 1;

    const int cbase = h * DHEAD + lane * 4;
    const size_t rowQ = (size_t)(b * SEQ + s) * QKVN;
    v4bf qv = *reinterpret_cast<const v4bf*>(qkv + rowQ + cbase);
    const float q0 = (float)qv[0], q1 = (float)qv[1];
    const float q2 = (float)qv[2], q3 = (float)qv[3];

    const float scale = 0.08838834764831845f;  // 1/sqrt(128)
    float logit[12];
#pragma unroll
    for (int a = 0; a < 12; ++a) {
        const size_t rowK = (size_t)(b * SEQ + anc[a]) * QKVN + DMODEL;
        v4bf kv = *reinterpret_cast<const v4bf*>(qkv + rowK + cbase);
        float p = q0 * (float)kv[0] + q1 * (float)kv[1] +
                  q2 * (float)kv[2] + q3 * (float)kv[3];
#pragma unroll
        for (int o = 16; o > 0; o >>= 1) p += __shfl_xor(p, o);
        const float lw = a < 6 ? lw0 : (a < 10 ? lw1 : lw2);
        logit[a] = p * scale + lw;
    }

    float m = logit[0];
#pragma unroll
    for (int a = 1; a < 12; ++a) m = fmaxf(m, logit[a]);
    float e[12], sum = 0.f;
#pragma unroll
    for (int a = 0; a < 12; ++a) { e[a] = __expf(logit[a] - m); sum += e[a]; }
    const float inv = 1.0f / sum;

    float c0 = 0.f, c1 = 0.f, c2 = 0.f, c3 = 0.f;
#pragma unroll
    for (int a = 0; a < 12; ++a) {
        const size_t rowV = (size_t)(b * SEQ + anc[a]) * QKVN + 2 * DMODEL;
        v4bf vv = *reinterpret_cast<const v4bf*>(qkv + rowV + cbase);
        const float w = e[a] * inv;
        c0 += w * (float)vv[0]; c1 += w * (float)vv[1];
        c2 += w * (float)vv[2]; c3 += w * (float)vv[3];
    }

    v4bf o;
    o[0] = (bf16)c0; o[1] = (bf16)c1; o[2] = (bf16)c2; o[3] = (bf16)c3;
    *reinterpret_cast<v4bf*>(ctx + (size_t)(b * SEQ + s) * DMODEL + cbase) = o;
}

// ---------------------------------------------------------------------------
// Launcher
// ---------------------------------------------------------------------------
extern "C" void kernel_launch(void* const* d_in, const int* in_sizes, int n_in,
                              void* d_out, int out_size, void* d_ws, size_t ws_size,
                              hipStream_t stream) {
    (void)in_sizes; (void)n_in; (void)out_size; (void)ws_size;
    const float* x      = (const float*)d_in[0];  // (2,4096,1536)
    const float* qkv_w  = (const float*)d_in[1];  // (4608,1536)
    const float* qkv_b  = (const float*)d_in[2];  // (4608,)
    const float* out_w  = (const float*)d_in[3];  // (1536,1536)
    const float* out_b  = (const float*)d_in[4];  // (1536,)
    const float* gscale = (const float*)d_in[5];  // (3,)
    float* out = (float*)d_out;                   // (2,4096,1536)

    // workspace layout (256B-aligned offsets)
    char* w = (char*)d_ws;
    bf16* x_bf   = (bf16*)(w);                                 //  25,165,824 B
    bf16* w1_bf  = (bf16*)(w + 25165824);                      //  14,155,776 B
    bf16* w2_bf  = (bf16*)(w + 39321600);                      //   4,718,592 B
    bf16* qkv_bf = (bf16*)(w + 44040192);                      //  75,497,472 B
    bf16* ctx_bf = (bf16*)(w + 119537664);                     //  25,165,824 B

    // 1) fp32 -> bf16 conversions
    {
        int n4 = (MROWS * DMODEL) / 4;   // x
        f32_to_bf16_x4<<<(n4 + 255) / 256, 256, 0, stream>>>(x, x_bf, n4);
        n4 = (QKVN * DMODEL) / 4;        // qkv_w
        f32_to_bf16_x4<<<(n4 + 255) / 256, 256, 0, stream>>>(qkv_w, w1_bf, n4);
        n4 = (DMODEL * DMODEL) / 4;      // out_w
        f32_to_bf16_x4<<<(n4 + 255) / 256, 256, 0, stream>>>(out_w, w2_bf, n4);
    }

    // 2) qkv = x @ qkv_w^T + qkv_b   (bf16 out)
    {
        dim3 grid(QKVN / 128, MROWS / 128);  // 36 x 64
        gemm_bf16_wmma<bf16><<<grid, 256, 0, stream>>>(
            x_bf, w1_bf, qkv_b, qkv_bf, MROWS, QKVN, DMODEL);
    }

    // 3) anchor attention -> ctx (bf16)
    {
        const int waves = BATCH * NHEAD * SEQ;   // 98304
        anchor_attn<<<waves / 8, 256, 0, stream>>>(qkv_bf, gscale, ctx_bf);
    }

    // 4) out = ctx @ out_w^T + out_b  (fp32 out)
    {
        dim3 grid(DMODEL / 128, MROWS / 128);    // 12 x 64
        gemm_bf16_wmma<float><<<grid, 256, 0, stream>>>(
            ctx_bf, w2_bf, out_b, out, MROWS, DMODEL, DMODEL);
    }
}
